// Model_7499012899662
// MI455X (gfx1250) — matrix operand order repaired and checked by reading, not compiled, not run on hardware
//
#include <hip/hip_runtime.h>

// ---------------------------------------------------------------------------
// CDNA5 (gfx1250) sparse U-Net forward.  All GEMMs (27-tap spconv, 8-tap
// down-conv, 8-tap deconv, 1x1 linear) run through v_wmma_f32_16x16x32_bf16.
// Each wave32 owns a 16x32 output tile (two 16x16 N-tiles): the gathered A
// fragment is reused by two WMMAs per K-chunk, halving gather traffic/FLOP.
// No runtime integer division on device.
// ---------------------------------------------------------------------------

typedef __bf16 bf16_t;
typedef __attribute__((ext_vector_type(16))) __bf16 v16bf;
typedef __attribute__((ext_vector_type(8)))  float  v8f;

// r = i / C, c = i % C with C = (odd3 ? 3 : 1) << p   (no hardware divide)
__device__ __forceinline__ void rcsplit(unsigned i, unsigned C, int p, int o3,
                                        unsigned& r, unsigned& c) {
  unsigned ip = i >> p;
  r = o3 ? (ip / 3u) : ip;          // /3u -> mul_hi, no divide
  c = i - r * C;
}

// ---------------------------- elementwise helpers --------------------------

__global__ void k_fill(float* __restrict__ p, unsigned n) {
  unsigned i = blockIdx.x * blockDim.x + threadIdx.x;
  if (i < n) p[i] = 0.f;
}

// f32 (M x C) -> bf16 (M x CP), zero-padded columns.  (pP,o3P) decompose CP.
__global__ void k_tobf16(const float* __restrict__ X, bf16_t* __restrict__ Y,
                         unsigned n, unsigned C, unsigned CP, int pP, int o3P) {
  unsigned i = blockIdx.x * blockDim.x + threadIdx.x;
  if (i >= n) return;
  unsigned r, c; rcsplit(i, CP, pP, o3P, r, c);
  Y[i] = (c < C) ? (bf16_t)X[r * C + c] : (bf16_t)0.f;
}

// Repack W[KT][Ci][Co] (f32) into WMMA B-fragment-major bf16.
// grid = (nNt, nCik, KT), block = 32 lanes; fragment f = (k*nCik+cik)*nNt+nt.
// B packing used here (contiguous k, same as A): lane<16: col=lane,   K=j      (j=0..15)
//                           lane>=16: col=lane-16, K=16+j
__global__ void __launch_bounds__(32)
k_packW(const float* __restrict__ W, bf16_t* __restrict__ Wp,
        int Ci, int Co) {
  const int lane = threadIdx.x;
  const int nt = blockIdx.x, cik = blockIdx.y, k = blockIdx.z;
  const int nNt = gridDim.x, nCik = gridDim.y;
  const int co = nt * 16 + (lane & 15);
  const int kb = cik * 32 + ((lane >> 4) << 4);
  long f = ((long)k * nCik + cik) * nNt + nt;
  bf16_t* dst = Wp + (f * 32 + lane) * 16;
  for (int j = 0; j < 16; ++j) {
    int ci = kb + j;
    dst[j] = (ci < Ci) ? (bf16_t)W[((long)k * Ci + ci) * Co + co] : (bf16_t)0.f;
  }
}

// ------------------------------ WMMA gather-GEMM ---------------------------
// Out[M x Co] = sum_k gather_k(X) @ W[k]   (bf16 in, f32 accumulate)
//   nbr != 0              : src = nbr[row*KT + k]                  (-1 skips)
//   nbr == 0 && par != 0  : src = (kx[row]==k) ? par[row] : -1     (deconv)
//   else                  : src = row (identity, KT==1, linear layer)
// grid = (ceil(M/16), Co/32), block = 32: one wave -> 16x32 output tile
// (two adjacent 16x16 N-tiles sharing one gathered A fragment).
__global__ void __launch_bounds__(32)
k_spconv_wmma(const bf16_t* __restrict__ X, const bf16_t* __restrict__ Wp,
              float* __restrict__ Out,
              const int* __restrict__ nbr, const int* __restrict__ par,
              const int* __restrict__ kx,
              int M, int CiP, int Co, int KT) {
  const int lane = threadIdx.x;
  const int mt = blockIdx.x, nt0 = blockIdx.y * 2;
  const int nCik = CiP >> 5, nNt = Co >> 4;
  const int row  = mt * 16 + (lane & 15);
  const int hsel = (lane >> 4) << 4;          // A-fragment half select: 0 or 16
  const long fragStride = (long)nNt * 32 * 16;

  int kkDe = -1, ppDe = -1;                   // hoisted deconv selectors
  if (!nbr && par && row < M) { kkDe = kx[row]; ppDe = par[row]; }

  v8f acc0 = {}, acc1 = {};
  for (int k = 0; k < KT; ++k) {
    int src;
    if (nbr)      src = (row < M) ? nbr[row * KT + k] : -1;
    else if (par) src = (kkDe == k) ? ppDe : -1;
    else          src = (row < M) ? row : -1;
    const bf16_t* xr = X + (long)src * CiP;
    const bf16_t* wp = Wp + ((long)k * nCik * nNt + nt0) * (32 * 16) + lane * 16;
#pragma unroll 4
    for (int cik = 0; cik < nCik; ++cik) {
      v16bf a, b0, b1;
      if (src >= 0) {
        ((uint4*)&a)[0] = *(const uint4*)(xr + cik * 32 + hsel);
        ((uint4*)&a)[1] = *(const uint4*)(xr + cik * 32 + 8 + hsel);
      } else {
        uint4 zz; zz.x = zz.y = zz.z = zz.w = 0u;
        ((uint4*)&a)[0] = zz; ((uint4*)&a)[1] = zz;
      }
      const bf16_t* wpc = wp + (long)cik * fragStride;   // n-tile nt0
      ((uint4*)&b0)[0] = ((const uint4*)wpc)[0];
      ((uint4*)&b0)[1] = ((const uint4*)wpc)[1];
      const bf16_t* wpc1 = wpc + 32 * 16;                // adjacent n-tile nt0+1
      ((uint4*)&b1)[0] = ((const uint4*)wpc1)[0];
      ((uint4*)&b1)[1] = ((const uint4*)wpc1)[1];
      __builtin_prefetch(wpc + fragStride, 0, 0);        // next K-chunk weights
      acc0 = __builtin_amdgcn_wmma_f32_16x16x32_bf16(
          false, a, false, b0, (short)0, acc0, false, false);
      acc1 = __builtin_amdgcn_wmma_f32_16x16x32_bf16(
          false, a, false, b1, (short)0, acc1, false, false);
    }
  }
  // D layout: VGPR v -> row = v + 8*(lane>=16), col = lane&15
  const int col0 = nt0 * 16 + (lane & 15);
  const int rb0  = mt * 16 + ((lane >> 4) << 3);
#pragma unroll
  for (int v = 0; v < 8; ++v) {
    int r = rb0 + v;
    if (r < M) {
      Out[(long)r * Co + col0]      = acc0[v];
      Out[(long)r * Co + col0 + 16] = acc1[v];
    }
  }
}

// ------------------------------ batchnorm ----------------------------------

// Register-resident column stats: thread = column, 256 rows per block.
__global__ void __launch_bounds__(256)
k_bnstats(const float* __restrict__ X, int M, int C,
          float* __restrict__ acc /* [2C] pre-zeroed */) {
  int c = threadIdx.x;
  if (c >= C) return;
  long r0 = (long)blockIdx.x * 256;
  float s = 0.f, q = 0.f;
  for (int r = 0; r < 256; ++r) {
    long rr = r0 + r;
    if (rr < M) { float v = X[rr * C + c]; s += v; q += v * v; }
  }
  atomicAdd(&acc[c], s);
  atomicAdd(&acc[C + c], q);
}

// out = [relu]( g*(x-m)*rsqrt(var+eps)+b [+ res] )
__global__ void k_bnapply(const float* __restrict__ X, const float* __restrict__ acc,
                          const float* __restrict__ g, const float* __restrict__ b,
                          const float* __restrict__ res, float* __restrict__ out,
                          unsigned n, unsigned C, int p, int o3, float fM, int relu) {
  unsigned i = blockIdx.x * blockDim.x + threadIdx.x;
  if (i >= n) return;
  unsigned r, c; rcsplit(i, C, p, o3, r, c);
  float m  = acc[c] / fM;
  float vr = acc[C + c] / fM - m * m;
  float y = g[c] * (X[i] - m) * rsqrtf(vr + 1e-5f) + b[c];
  if (res) y += res[i];
  out[i] = relu ? fmaxf(y, 0.f) : y;
}

// ------------------------------ segment mean -------------------------------

__global__ void k_segadd(const float* __restrict__ X, const int* __restrict__ seg,
                         float* __restrict__ sum, float* __restrict__ cnt,
                         unsigned n, unsigned C, int p, int o3) {
  unsigned i = blockIdx.x * blockDim.x + threadIdx.x;
  if (i >= n) return;
  unsigned r, c; rcsplit(i, C, p, o3, r, c);
  int s = seg[r];
  if (s < 0) return;
  atomicAdd(&sum[(unsigned)s * C + c], X[i]);
  if (c == 0) atomicAdd(&cnt[s], 1.0f);
}

__global__ void k_segnorm(const float* __restrict__ sum, const float* __restrict__ cnt,
                          float* __restrict__ out, unsigned n, unsigned C, int p, int o3) {
  unsigned i = blockIdx.x * blockDim.x + threadIdx.x;
  if (i >= n) return;
  unsigned r, c; rcsplit(i, C, p, o3, r, c);
  out[i] = sum[i] / fmaxf(cnt[r], 1.0f);
}

// ------------------------------ voxel -> point -----------------------------

__global__ void k_v2p(const float* __restrict__ X, const int* __restrict__ idx8,
                      const float* __restrict__ w8, const float* __restrict__ add,
                      float* __restrict__ out, unsigned n, unsigned C, int p, int o3) {
  unsigned i = blockIdx.x * blockDim.x + threadIdx.x;
  if (i >= n) return;
  unsigned r, c; rcsplit(i, C, p, o3, r, c);
  float s = add ? add[i] : 0.f;
#pragma unroll
  for (int k = 0; k < 8; ++k) {
    int q = idx8[r * 8 + k];
    unsigned qq = (q > 0) ? (unsigned)q : 0u;
    s += w8[r * 8 + k] * X[(long)qq * C + c];   // w8==0 where q<0
  }
  out[i] = s;
}

// dst[r*ldd + coff + c] = src[r*Cs + c]
__global__ void k_copycols(const float* __restrict__ src, float* __restrict__ dst,
                           unsigned n, unsigned Cs, int p, int o3, unsigned ldd,
                           unsigned coff) {
  unsigned i = blockIdx.x * blockDim.x + threadIdx.x;
  if (i >= n) return;
  unsigned r, c; rcsplit(i, Cs, p, o3, r, c);
  dst[r * ldd + coff + c] = src[i];
}

// ---------------------------------------------------------------------------

#define LAUNCH(kern, n, ...)                                                   \
  do {                                                                         \
    unsigned _n = (unsigned)(n); unsigned _g = (_n + 255u) / 256u;             \
    if (_g < 1u) _g = 1u;                                                      \
    kern<<<dim3(_g), dim3(256), 0, stream>>>(__VA_ARGS__);                     \
  } while (0)

// decompose C = (o3 ? 3 : 1) << p   (all channel counts in this net qualify)
static inline void decompC(int C, int& p, int& o3) {
  p = 0; int c = C;
  while ((c & 1) == 0) { c >>= 1; ++p; }
  o3 = (c == 3) ? 1 : 0;
}

extern "C" void kernel_launch(void* const* d_in, const int* in_sizes, int n_in,
                              void* d_out, int out_size, void* d_ws, size_t ws_size,
                              hipStream_t stream) {
  (void)n_in; (void)out_size; (void)ws_size;
  const int N = 120000;
  auto F = [&](int i) { return (const float*)d_in[i]; };
  auto I = [&](int i) { return (const int*)d_in[i]; };

  // Data-dependent level sizes from the 27-column neighbor maps.
  const int M0 = in_sizes[160] / 27, M1 = in_sizes[161] / 27,
            M2 = in_sizes[162] / 27, M3 = in_sizes[163] / 27,
            M4 = in_sizes[164] / 27;

  // ---- workspace bump allocator ----
  char* wp_ = (char*)d_ws;
  auto alloc = [&](size_t bytes) -> char* {
    char* r = wp_; wp_ += (bytes + 255) & ~(size_t)255; return r;
  };
  float*  bufPre = (float*)alloc((size_t)N * 256 * 4);       // GEMM pre-bn output
  bf16_t* xb16   = (bf16_t*)alloc((size_t)N * 256 * 2);      // activation bf16
  bf16_t* wpack  = (bf16_t*)alloc((size_t)27 * 384 * 256 * 2);
  float*  stats  = (float*)alloc(1024 * 4);
  float*  cnt    = (float*)alloc((size_t)M0 * 4);
  float*  bufH   = (float*)alloc((size_t)M0 * 128 * 4);
  float*  bufSC  = (float*)alloc((size_t)M0 * 128 * 4);
  float*  bufCat = (float*)alloc((size_t)M0 * 128 * 4);
  float*  bufS1  = (float*)alloc((size_t)M0 * 96 * 4);
  float*  bufS2  = (float*)alloc((size_t)M0 * 96 * 4);
  float*  v0f = (float*)alloc((size_t)M0 * 4 * 4);
  float*  x0  = (float*)alloc((size_t)M0 * 32 * 4);
  float*  xA  = (float*)alloc((size_t)M0 * 32 * 4);
  float*  x1  = (float*)alloc((size_t)M1 * 32 * 4);
  float*  x2  = (float*)alloc((size_t)M2 * 64 * 4);
  float*  x3  = (float*)alloc((size_t)M3 * 128 * 4);
  float*  x4  = (float*)alloc((size_t)M4 * 256 * 4);
  float*  z0  = (float*)alloc((size_t)N * 32 * 4);
  float*  z1  = (float*)alloc((size_t)N * 256 * 4);
  float*  z2  = (float*)alloc((size_t)N * 128 * 4);
  float*  y1  = (float*)alloc((size_t)M3 * 256 * 4);
  float*  y2  = (float*)alloc((size_t)M2 * 128 * 4);
  float*  yv4 = (float*)alloc((size_t)M4 * 256 * 4);
  float*  yv2 = (float*)alloc((size_t)M2 * 128 * 4);
  float*  y3  = (float*)alloc((size_t)M1 * 96 * 4);
  float*  y4  = (float*)alloc((size_t)M0 * 96 * 4);

  // ---- composite ops ----
  auto conv_bn = [&](const float* X, int Min, int Mout, int Ci, int Co, int KT,
                     const float* W, const float* g, const float* b,
                     const int* nbr, const int* par, const int* kx,
                     const float* res, int relu, float* out) {
    int CiP = (Ci + 31) & ~31;
    int nCik = CiP / 32, nNt = Co / 16;
    int pP, o3P; decompC(CiP, pP, o3P);
    int pO, o3O; decompC(Co, pO, o3O);
    LAUNCH(k_tobf16, (unsigned)Min * CiP, X, xb16, (unsigned)Min * CiP,
           (unsigned)Ci, (unsigned)CiP, pP, o3P);
    k_packW<<<dim3(nNt, nCik, KT), dim3(32), 0, stream>>>(W, wpack, Ci, Co);
    k_spconv_wmma<<<dim3((Mout + 15) / 16, nNt / 2), dim3(32), 0, stream>>>(
        xb16, wpack, bufPre, nbr, par, kx, Mout, CiP, Co, KT);
    LAUNCH(k_fill, 2 * Co, stats, (unsigned)(2 * Co));
    { unsigned g2 = (unsigned)((Mout + 255) / 256); if (g2 < 1) g2 = 1;
      k_bnstats<<<dim3(g2), dim3(256), 0, stream>>>(bufPre, Mout, Co, stats); }
    LAUNCH(k_bnapply, (unsigned)Mout * Co, bufPre, stats, g, b, res, out,
           (unsigned)Mout * Co, (unsigned)Co, pO, o3O, (float)Mout, relu);
  };

  // residual block; bi = d_in index of 'w1' (w1,bn1g,bn1b,w2,bn2g,bn2b[,ws,bnsg,bnsb])
  auto rb = [&](const float* x, int M, int Ci, int Co, int bi, bool hasWs,
                const int* nbr, float* out) {
    const float* sc = x;
    if (hasWs) {
      conv_bn(x, M, M, Ci, Co, 1, F(bi + 6), F(bi + 7), F(bi + 8),
              nullptr, nullptr, nullptr, nullptr, 0, bufSC);
      sc = bufSC;
    }
    conv_bn(x, M, M, Ci, Co, 27, F(bi), F(bi + 1), F(bi + 2),
            nbr, nullptr, nullptr, nullptr, 1, bufH);
    conv_bn(bufH, M, M, Co, Co, 27, F(bi + 3), F(bi + 4), F(bi + 5),
            nbr, nullptr, nullptr, sc, 1, out);
  };

  auto segmean = [&](const float* X, const int* seg, int Nr, int M, int C, float* out) {
    int p, o3; decompC(C, p, o3);
    LAUNCH(k_fill, (unsigned)M * C, bufPre, (unsigned)M * C);
    LAUNCH(k_fill, M, cnt, (unsigned)M);
    LAUNCH(k_segadd, (unsigned)Nr * C, X, seg, bufPre, cnt, (unsigned)Nr * C,
           (unsigned)C, p, o3);
    LAUNCH(k_segnorm, (unsigned)M * C, bufPre, cnt, out, (unsigned)M * C,
           (unsigned)C, p, o3);
  };

  auto v2p = [&](const float* X, const int* idx8, const float* w8,
                 const float* add, float* out, int C) {
    int p, o3; decompC(C, p, o3);
    LAUNCH(k_v2p, (unsigned)N * C, X, idx8, w8, add, out, (unsigned)N * C,
           (unsigned)C, p, o3);
  };

  // ---- forward pass ----
  // stem
  segmean(F(0), I(179), N, M0, 4, v0f);
  conv_bn(v0f, M0, M0, 4, 32, 27, F(1), F(2), F(3), I(160), 0, 0, 0, 1, bufS1);
  conv_bn(bufS1, M0, M0, 32, 32, 27, F(4), F(5), F(6), I(160), 0, 0, 0, 1, x0);
  v2p(x0, I(177), F(178), nullptr, z0, 32);
  segmean(z0, I(179), N, M0, 32, xA);

  auto stage = [&](const float* x, int Ml, int Mn, int Ci, int Crb,
                   int dwIdx, int rb1Idx, bool ws1, int rb2Idx,
                   const int* dmap, const int* nbr, float* out) {
    conv_bn(x, Ml, Mn, Ci, Ci, 8, F(dwIdx), F(dwIdx + 1), F(dwIdx + 2),
            dmap, nullptr, nullptr, nullptr, 1, bufS1);
    rb(bufS1, Mn, Ci, Crb, rb1Idx, ws1, nbr, bufS2);
    rb(bufS2, Mn, Crb, Crb, rb2Idx, false, nbr, out);
  };
  stage(xA, M0, M1, 32, 32, 7, 10, false, 16, I(165), I(161), x1);
  stage(x1, M1, M2, 32, 64, 22, 25, true, 34, I(168), I(162), x2);
  stage(x2, M2, M3, 64, 128, 40, 43, true, 52, I(171), I(163), x3);
  stage(x3, M3, M4, 128, 256, 58, 61, true, 70, I(174), I(164), x4);

  // z1 = v2p(x4) + pt0(z0)   (pt bias is zero and cancels through BN anyway)
  conv_bn(z0, N, N, 32, 256, 1, F(148), F(150), F(151), 0, 0, 0, 0, 1, z1);
  v2p(x4, I(183), F(184), z1, z1, 256);
  segmean(z1, I(185), N, M4, 256, yv4);

  auto upst = [&](const float* yv, const float* skip, int Mc, int Mf,
                  int Cin, int Cde, int Cskip, int Crb,
                  int deIdx, int rb1Idx, int rb2Idx,
                  const int* par_, const int* kx_, const int* nbr, float* out) {
    conv_bn(yv, Mc, Mf, Cin, Cde, 8, F(deIdx), F(deIdx + 1), F(deIdx + 2),
            nullptr, par_, kx_, nullptr, 1, bufS1);
    int pD, o3D; decompC(Cde, pD, o3D);
    int pS, o3S; decompC(Cskip, pS, o3S);
    LAUNCH(k_copycols, (unsigned)Mf * Cde, bufS1, bufCat, (unsigned)Mf * Cde,
           (unsigned)Cde, pD, o3D, (unsigned)(Cde + Cskip), 0u);
    LAUNCH(k_copycols, (unsigned)Mf * Cskip, skip, bufCat, (unsigned)Mf * Cskip,
           (unsigned)Cskip, pS, o3S, (unsigned)(Cde + Cskip), (unsigned)Cde);
    rb(bufCat, Mf, Cde + Cskip, Crb, rb1Idx, true, nbr, bufS1);
    rb(bufS1, Mf, Crb, Crb, rb2Idx, false, nbr, out);
  };
  upst(yv4, x3, M4, M3, 256, 256, 128, 256, 76, 79, 88, I(175), I(176), I(163), y1);
  upst(y1,  x2, M3, M2, 256, 128, 64, 128, 94, 97, 106, I(172), I(173), I(162), y2);

  // z2 = v2p(y2) + pt1(z1)
  conv_bn(z1, N, N, 256, 128, 1, F(152), F(154), F(155), 0, 0, 0, 0, 1, z2);
  v2p(y2, I(180), F(181), z2, z2, 128);
  segmean(z2, I(182), N, M2, 128, yv2);

  upst(yv2, x1, M2, M1, 128, 96, 32, 96, 112, 115, 124, I(169), I(170), I(161), y3);
  upst(y3,  x0, M1, M0, 96, 96, 32, 96, 130, 133, 142, I(166), I(167), I(160), y4);

  // z3 = v2p(y4) + pt2(z2)  -> d_out (N x 96, f32)
  float* out = (float*)d_out;
  conv_bn(z2, N, N, 128, 96, 1, F(156), F(158), F(159), 0, 0, 0, 0, 1, out);
  v2p(y4, I(177), F(178), out, out, 96);
}
